// MoEFFNLayer_56684978372821
// MI455X (gfx1250) — matrix operand order repaired and checked by reading, not compiled, not run on hardware
//
#include <hip/hip_runtime.h>
#include <hip/hip_bf16.h>
#include <math.h>

// Problem constants (B=2, S=2048, D=1024, F=4096, E=8)
#define TOKS 4096
#define DDIM 1024
#define FDIM 4096
#define NEXP 8
#define FFN_GRID_Y 64

// FFN tile: 32 tokens. LDS = x(32x1024 bf16) + h(32x4096 bf16) = 320 KB exactly.
// Bank conflicts avoided with a 16B XOR swizzle instead of padding:
//   addr = base + row*stride + (off ^ ((row&15)<<4))
#define XSB 2048                    // x row stride (bytes)
#define HSB 8192                    // h row stride (bytes)
#define HS_BASE (32 * XSB)          // 65536
#define SMEM_BYTES (32 * XSB + 32 * HSB)  // 327680 = 320 KB

typedef __attribute__((ext_vector_type(16))) __bf16 v16bf;
typedef __attribute__((ext_vector_type(8)))  float  v8f;

union ABReg {
  v16bf v;
  uint4 q[2];
  unsigned int u[8];
};

// ---- f32 -> bf16 (RNE) -----------------------------------------------------
__device__ __forceinline__ unsigned int f2bf_manual(float f) {
  unsigned int u = __float_as_uint(f);
  u += 0x7FFFu + ((u >> 16) & 1u);
  return u >> 16;
}

#if defined(__has_builtin)
#if __has_builtin(__builtin_amdgcn_cvt_pk_bf16_f32)
#define HAVE_CVT_PK_BF16 1
#endif
#endif

#ifdef HAVE_CVT_PK_BF16
typedef __attribute__((ext_vector_type(2))) __bf16 v2bf;
__device__ __forceinline__ unsigned int pk2bf(float lo, float hi) {
  union { v2bf v; unsigned int u; } c;
  c.v = __builtin_amdgcn_cvt_pk_bf16_f32(lo, hi);
  return c.u;
}
#else
__device__ __forceinline__ unsigned int pk2bf(float lo, float hi) {
  return f2bf_manual(lo) | (f2bf_manual(hi) << 16);
}
#endif

__device__ __forceinline__ unsigned int swz16(unsigned int off, int row) {
  return off ^ (unsigned int)((row & 15) << 4);
}

__device__ __forceinline__ v8f wmma_bf16(const ABReg& a, const ABReg& b, v8f c) {
  // D = A(16x32 bf16) * B(32x16 bf16) + C(16x16 f32)
  return __builtin_amdgcn_wmma_f32_16x16x32_bf16(
      /*neg_a=*/false, a.v, /*neg_b=*/false, b.v,
      /*c_mod=*/(short)0, c, /*reuse_a=*/false, /*reuse_b=*/false);
}

// ---------------------------------------------------------------------------
// Kernel 0: zero the tgt2 accumulator (atomic target) and expert counts.
// ---------------------------------------------------------------------------
__global__ __launch_bounds__(256) void moe_zero_kernel(float4* __restrict__ accum,
                                                       int* __restrict__ counts) {
  size_t i = (size_t)blockIdx.x * blockDim.x + threadIdx.x;  // TOKS*DDIM/4 threads
  accum[i] = make_float4(0.f, 0.f, 0.f, 0.f);
  if (i < NEXP) counts[i] = 0;
}

// ---------------------------------------------------------------------------
// Kernel 0b: stream-convert a weight tensor f32 -> bf16 (8 elems / thread).
// ---------------------------------------------------------------------------
__global__ __launch_bounds__(256) void moe_w2bf_kernel(const float4* __restrict__ src,
                                                       uint4* __restrict__ dst) {
  const size_t i = (size_t)blockIdx.x * blockDim.x + threadIdx.x;
  const float4 a = src[2 * i];
  const float4 b = src[2 * i + 1];
  uint4 p;
  p.x = pk2bf(a.x, a.y);
  p.y = pk2bf(a.z, a.w);
  p.z = pk2bf(b.x, b.y);
  p.w = pk2bf(b.z, b.w);
  dst[i] = p;
}

// ---------------------------------------------------------------------------
// Kernel 1: router. One wave per token: logits[8] = x . Wr[e] + br[e],
// wave-reduce, top-2 softmax, scatter (token, weight) into per-expert lists.
// ---------------------------------------------------------------------------
__global__ __launch_bounds__(256) void moe_router_kernel(
    const float* __restrict__ tgt, const float* __restrict__ Wr,
    const float* __restrict__ br, int* __restrict__ counts,
    int* __restrict__ lists, float* __restrict__ wlists) {
  const int wv = threadIdx.x >> 5;
  const int lane = threadIdx.x & 31;
  const int t = blockIdx.x * 8 + wv;
  const float* x = tgt + (size_t)t * DDIM;

  float acc[NEXP];
#pragma unroll
  for (int e = 0; e < NEXP; ++e) acc[e] = 0.0f;
  for (int d = lane; d < DDIM; d += 32) {
    const float xv = x[d];
#pragma unroll
    for (int e = 0; e < NEXP; ++e) acc[e] = fmaf(xv, Wr[e * DDIM + d], acc[e]);
  }
#pragma unroll
  for (int e = 0; e < NEXP; ++e) {
#pragma unroll
    for (int off = 16; off; off >>= 1) acc[e] += __shfl_xor(acc[e], off, 32);
  }

  if (lane == 0) {
    float l[NEXP];
#pragma unroll
    for (int e = 0; e < NEXP; ++e) l[e] = acc[e] + br[e];
    int i1 = 0; float v1 = l[0];
#pragma unroll
    for (int e = 1; e < NEXP; ++e) if (l[e] > v1) { v1 = l[e]; i1 = e; }
    int i2 = (i1 == 0) ? 1 : 0; float v2 = l[i2];
#pragma unroll
    for (int e = 0; e < NEXP; ++e) if (e != i1 && l[e] > v2) { v2 = l[e]; i2 = e; }
    const float bexp = __expf(v2 - v1);
    const float w1 = 1.0f / (1.0f + bexp);
    const float w2 = 1.0f - w1;
    int p1 = atomicAdd(&counts[i1], 1);
    lists[i1 * TOKS + p1] = t;  wlists[i1 * TOKS + p1] = w1;
    int p2 = atomicAdd(&counts[i2], 1);
    lists[i2 * TOKS + p2] = t;  wlists[i2 * TOKS + p2] = w2;
  }
}

// ---------------------------------------------------------------------------
// Kernel 2: expert FFN over gathered 32-token tiles (weights already bf16).
// Operand roles: A = weight tile (global bf16), B = activation tile (LDS),
// so D rows = output features, D cols = tokens:
//   - each lane's 8 results = 8 consecutive features of ONE token row
//     -> epilogues are vectorized (1x ds_store_b128 / 8 contiguous atomics)
//   - B (activations) is shared across the 4 grouped feature-tiles:
//     4x ds_load_b128 + 8x global_load_b128 feed 8 WMMAs per k-step.
// LDS = 320 KB exactly (CDNA5 per-workgroup max), XOR-swizzled.
// ---------------------------------------------------------------------------
__global__ __launch_bounds__(256, 1) void moe_ffn_kernel(
    const float* __restrict__ tgt,
    const unsigned short* __restrict__ W1bf,
    const float* __restrict__ b1,
    const unsigned short* __restrict__ W2bf,
    const float* __restrict__ b2, const int* __restrict__ counts,
    const int* __restrict__ lists, const float* __restrict__ wlists,
    float* __restrict__ accum) {
  extern __shared__ char smem[];   // [0,64K): x tile / metadata, [64K,320K): h tile

  const int e = blockIdx.x;
  const int cnt = counts[e];
  const int ntile = (cnt + 31) >> 5;
  const unsigned short* W1e = W1bf + (size_t)e * FDIM * DDIM;
  const unsigned short* W2e = W2bf + (size_t)e * DDIM * FDIM;
  const float* b1e = b1 + e * FDIM;
  const float* b2e = b2 + e * DDIM;

  const int tid  = threadIdx.x;
  const int lane = tid & 31;
  const int wv   = tid >> 5;
  const int lrow = lane & 15;   // A feature row / B,C token column owned by lane
  const int lhi  = lane >> 4;   // half-wave (selects K range / C row half)

  for (int tile = blockIdx.y; tile < ntile; tile += gridDim.y) {
    // Gather x tile (32 tokens x 1024) f32 -> bf16 into LDS (swizzled rows).
    {
      const int r = tid >> 3;        // token row 0..31
      const int seg = tid & 7;       // 128-element segment
      const int idx = tile * 32 + r;
      const int tok = (idx < cnt) ? lists[e * TOKS + idx] : -1;
      const float4* src =
          (const float4*)(tgt + (size_t)(tok < 0 ? 0 : tok) * DDIM + seg * 128);
      char* rowb = smem + r * XSB;
#pragma unroll
      for (int j = 0; j < 16; ++j) {
        const float4 f0 = (tok >= 0) ? src[2 * j]     : make_float4(0, 0, 0, 0);
        const float4 f1 = (tok >= 0) ? src[2 * j + 1] : make_float4(0, 0, 0, 0);
        uint4 p;
        p.x = pk2bf(f0.x, f0.y); p.y = pk2bf(f0.z, f0.w);
        p.z = pk2bf(f1.x, f1.y); p.w = pk2bf(f1.z, f1.w);
        *(uint4*)(rowb + swz16(seg * 256 + j * 16, r)) = p;
      }
    }
    __syncthreads();

    // ---- up-projection: h = relu(W1 x^T + b1), 8 groups of 4 f-tiles ----
    {
      const char* xrow0 = smem + lrow * XSB;          // token lrow (cols 0-15)
      const char* xrow1 = smem + (lrow + 16) * XSB;   // token lrow+16
      char* hb = smem + HS_BASE;
      for (int ntg = 0; ntg < 8; ++ntg) {
        const int f0tile = wv * 512 + ntg * 64;       // 4 feature-tiles of 16
        const uint4* wpa[4];
#pragma unroll
        for (int i = 0; i < 4; ++i)
          wpa[i] = (const uint4*)(W1e + (size_t)(f0tile + i * 16 + lrow) * DDIM);
        v8f acc[4][2] = {};
        for (int k0 = 0; k0 < DDIM; k0 += 32) {
          const unsigned int boff = (unsigned int)(2 * k0 + lhi * 32);
          ABReg bx0, bx1;
          // B 32x16 bf16: lane&15 = token col, half-wave picks K 0-15 / 16-31
          bx0.q[0] = *(const uint4*)(xrow0 + swz16(boff, lrow));
          bx0.q[1] = *(const uint4*)(xrow0 + swz16(boff + 16, lrow));
          bx1.q[0] = *(const uint4*)(xrow1 + swz16(boff, lrow));
          bx1.q[1] = *(const uint4*)(xrow1 + swz16(boff + 16, lrow));
          const int qi = (k0 >> 3) + 2 * lhi;
#pragma unroll
          for (int i = 0; i < 4; ++i) {
            ABReg aw;
            // A 16x32 bf16: lane&15 = feature row; half-wave picks K 0-15 / 16-31
            aw.q[0] = wpa[i][qi];
            aw.q[1] = wpa[i][qi + 1];
            acc[i][0] = wmma_bf16(aw, bx0, acc[i][0]);
            acc[i][1] = wmma_bf16(aw, bx1, acc[i][1]);
          }
        }
        // Epilogue: C rows = features (m=v+8*lhi), cols = token (lane&15).
        // Lane's 8 values = 8 consecutive f of one token row -> 1 b128 store.
#pragma unroll
        for (int i = 0; i < 4; ++i) {
          const int fbase = f0tile + i * 16 + 8 * lhi;
          const float4 bb0 = *(const float4*)(b1e + fbase);
          const float4 bb1 = *(const float4*)(b1e + fbase + 4);
          const float ba[8] = {bb0.x, bb0.y, bb0.z, bb0.w,
                               bb1.x, bb1.y, bb1.z, bb1.w};
          const unsigned int hoff = (unsigned int)(fbase * 2);
#pragma unroll
          for (int g = 0; g < 2; ++g) {
            const int token = lrow + 16 * g;
            float hv[8];
#pragma unroll
            for (int j = 0; j < 8; ++j) {
              float h = acc[i][g][j] + ba[j];
              hv[j] = h > 0.0f ? h : 0.0f;
            }
            uint4 P;
            P.x = pk2bf(hv[0], hv[1]); P.y = pk2bf(hv[2], hv[3]);
            P.z = pk2bf(hv[4], hv[5]); P.w = pk2bf(hv[6], hv[7]);
            *(uint4*)(hb + token * HSB + swz16(hoff, token)) = P;
          }
        }
      }
    }
    __syncthreads();

    // x region is dead now: stash routing metadata there for the epilogue.
    if (tid < 32) {
      const int idx = tile * 32 + tid;
      ((int*)smem)[tid] = (idx < cnt) ? lists[e * TOKS + idx] : -1;
      ((float*)(smem + 128))[tid] = (idx < cnt) ? wlists[e * TOKS + idx] : 0.0f;
    }
    __syncthreads();

    // Per-lane routing scalars (token column = lane&15 / +16), loop-invariant.
    const int   tok0 = ((const int*)smem)[lrow];
    const int   tok1 = ((const int*)smem)[lrow + 16];
    const float wt0  = ((const float*)(smem + 128))[lrow];
    const float wt1  = ((const float*)(smem + 128))[lrow + 16];

    // ---- down-projection: o = W2 h^T + b2, 2 groups of 4 d-tiles ----
    {
      const char* hrow0 = smem + HS_BASE + lrow * HSB;
      const char* hrow1 = smem + HS_BASE + (lrow + 16) * HSB;
      for (int ntg = 0; ntg < 2; ++ntg) {
        const int d0tile = wv * 128 + ntg * 64;
        const uint4* wpa[4];
#pragma unroll
        for (int i = 0; i < 4; ++i)
          wpa[i] = (const uint4*)(W2e + (size_t)(d0tile + i * 16 + lrow) * FDIM);
        v8f acc[4][2] = {};
        for (int k0 = 0; k0 < FDIM; k0 += 32) {
          const unsigned int boff = (unsigned int)(2 * k0 + lhi * 32);
          ABReg bh0, bh1;
          bh0.q[0] = *(const uint4*)(hrow0 + swz16(boff, lrow));
          bh0.q[1] = *(const uint4*)(hrow0 + swz16(boff + 16, lrow));
          bh1.q[0] = *(const uint4*)(hrow1 + swz16(boff, lrow));
          bh1.q[1] = *(const uint4*)(hrow1 + swz16(boff + 16, lrow));
          const int qi = (k0 >> 3) + 2 * lhi;
#pragma unroll
          for (int i = 0; i < 4; ++i) {
            ABReg aw;
            aw.q[0] = wpa[i][qi];
            aw.q[1] = wpa[i][qi + 1];
            acc[i][0] = wmma_bf16(aw, bh0, acc[i][0]);
            acc[i][1] = wmma_bf16(aw, bh1, acc[i][1]);
          }
        }
        // Epilogue: lane's 8 values = 8 consecutive d of one token row.
#pragma unroll
        for (int i = 0; i < 4; ++i) {
          const int dbase = d0tile + i * 16 + 8 * lhi;
          const float4 c0 = *(const float4*)(b2e + dbase);
          const float4 c1 = *(const float4*)(b2e + dbase + 4);
          const float bv[8] = {c0.x, c0.y, c0.z, c0.w, c1.x, c1.y, c1.z, c1.w};
          if (tok0 >= 0) {
            float* dstp = accum + (size_t)tok0 * DDIM + dbase;
#pragma unroll
            for (int j = 0; j < 8; ++j)
              atomicAdd(dstp + j, (acc[i][0][j] + bv[j]) * wt0);
          }
          if (tok1 >= 0) {
            float* dstp = accum + (size_t)tok1 * DDIM + dbase;
#pragma unroll
            for (int j = 0; j < 8; ++j)
              atomicAdd(dstp + j, (acc[i][1][j] + bv[j]) * wt1);
          }
        }
      }
    }
    __syncthreads();
  }
}

// ---------------------------------------------------------------------------
// Kernel 3: residual + LayerNorm. One block (256 threads) per token.
// ---------------------------------------------------------------------------
__global__ __launch_bounds__(256) void moe_ln_kernel(
    const float* __restrict__ tgt, const float* __restrict__ accum,
    const float* __restrict__ gamma, const float* __restrict__ beta,
    float* __restrict__ out) {
  __shared__ float sred[16];
  const int t = blockIdx.x;
  const int tid = threadIdx.x;
  float y[4];
  float s = 0.0f, ss = 0.0f;
#pragma unroll
  for (int i = 0; i < 4; ++i) {
    const int d = tid + 256 * i;
    const float v = tgt[(size_t)t * DDIM + d] + accum[(size_t)t * DDIM + d];
    y[i] = v; s += v; ss += v * v;
  }
#pragma unroll
  for (int off = 16; off; off >>= 1) {
    s  += __shfl_xor(s, off, 32);
    ss += __shfl_xor(ss, off, 32);
  }
  const int wv = tid >> 5, lane = tid & 31;
  if (lane == 0) { sred[wv] = s; sred[8 + wv] = ss; }
  __syncthreads();
  if (tid == 0) {
    float ts = 0.0f, tss = 0.0f;
    for (int i = 0; i < 8; ++i) { ts += sred[i]; tss += sred[8 + i]; }
    const float mu = ts / (float)DDIM;
    const float var = tss / (float)DDIM - mu * mu;
    sred[0] = mu;
    sred[1] = rsqrtf(var + 1e-5f);
  }
  __syncthreads();
  const float mu = sred[0], rs = sred[1];
#pragma unroll
  for (int i = 0; i < 4; ++i) {
    const int d = tid + 256 * i;
    out[(size_t)t * DDIM + d] = (y[i] - mu) * rs * gamma[d] + beta[d];
  }
}

// ---------------------------------------------------------------------------
// Host launcher.
// Workspace layout (~151.3 MB):
//   [0, 64 MB)          bf16 W1bf[E][F][D]
//   [64 MB, 128 MB)     bf16 W2bf[E][D][F]
//   [128 MB, +16 MB)    float accum[TOKS*DDIM]   (tgt2 atomic accumulator)
//   [+0, +128)          int counts[8] (padded)
//   [..., +128K)        int lists[8][TOKS]
//   [..., +128K)        float wlists[8][TOKS]
// ---------------------------------------------------------------------------
extern "C" void kernel_launch(void* const* d_in, const int* in_sizes, int n_in,
                              void* d_out, int out_size, void* d_ws, size_t ws_size,
                              hipStream_t stream) {
  const float* tgt   = (const float*)d_in[0];
  const float* Wr    = (const float*)d_in[1];
  const float* br    = (const float*)d_in[2];
  const float* W1    = (const float*)d_in[3];
  const float* b1    = (const float*)d_in[4];
  const float* W2    = (const float*)d_in[5];
  const float* b2    = (const float*)d_in[6];
  const float* gamma = (const float*)d_in[7];
  const float* beta  = (const float*)d_in[8];
  float* out = (float*)d_out;

  const size_t wElems = (size_t)NEXP * FDIM * DDIM;          // 33,554,432
  const size_t wbfBytes = wElems * 2;                        // 64 MB each

  char* ws = (char*)d_ws;
  unsigned short* W1bf = (unsigned short*)ws;
  unsigned short* W2bf = (unsigned short*)(ws + wbfBytes);
  float* accum  = (float*)(ws + 2 * wbfBytes);
  const size_t accum_bytes = (size_t)TOKS * DDIM * sizeof(float);
  char* meta = ws + 2 * wbfBytes + accum_bytes;
  int*   counts = (int*)meta;
  int*   lists  = (int*)(meta + 128);
  float* wlists = (float*)(meta + 128 + (size_t)NEXP * TOKS * sizeof(int));

  (void)hipFuncSetAttribute(reinterpret_cast<const void*>(moe_ffn_kernel),
                            hipFuncAttributeMaxDynamicSharedMemorySize, SMEM_BYTES);

  moe_zero_kernel<<<(TOKS * DDIM / 4) / 256, 256, 0, stream>>>((float4*)accum, counts);
  const int convBlocks = (int)((wElems / 8) / 256);           // 16384
  moe_w2bf_kernel<<<convBlocks, 256, 0, stream>>>((const float4*)W1, (uint4*)W1bf);
  moe_w2bf_kernel<<<convBlocks, 256, 0, stream>>>((const float4*)W2, (uint4*)W2bf);
  moe_router_kernel<<<TOKS / 8, 256, 0, stream>>>(tgt, Wr, br, counts, lists, wlists);
  moe_ffn_kernel<<<dim3(NEXP, FFN_GRID_Y), 256, SMEM_BYTES, stream>>>(
      tgt, W1bf, b1, W2bf, b2, counts, lists, wlists, accum);
  moe_ln_kernel<<<TOKS, 256, 0, stream>>>(tgt, accum, gamma, beta, out);
}
